// GraphAttention_67465346285784
// MI455X (gfx1250) — matrix operand order repaired and checked by reading, not compiled, not run on hardware
//
#include <hip/hip_runtime.h>
#include <hip/hip_bf16.h>
#include <math.h>

#define B  32
#define N  1024
#define NI 256
#define NF 256
#define H  4

typedef __attribute__((ext_vector_type(16))) _Float16 v16h;
typedef __attribute__((ext_vector_type(8)))  _Float16 v8h;
typedef __attribute__((ext_vector_type(8)))  float    v8f;

union AF {
    v16h v;
    struct { v8h lo, hi; } p;
    _Float16 e[16];
};
union HF8 {
    v8h v;
    _Float16 h[8];
};

// ---------------------------------------------------------------------------
// Convert inp (f32) -> f16
__global__ void cvt_x_kernel(const float* __restrict__ x, _Float16* __restrict__ xh) {
    int idx = blockIdx.x * 256 + threadIdx.x;   // over B*N*NI
    xh[idx] = (_Float16)x[idx];
}

// Convert Wk [H][NI][NF] (f32) -> Wt [H][NF][NI] (f16, transposed)
__global__ void cvt_w_kernel(const float* __restrict__ Wk, _Float16* __restrict__ Wt) {
    int idx = blockIdx.x * 256 + threadIdx.x;   // over H*NF*NI
    int i = idx & (NI - 1);
    int f = (idx >> 8) & (NF - 1);
    int h = idx >> 16;
    Wt[idx] = (_Float16)Wk[(h * NI + i) * NF + f];
}

// ---------------------------------------------------------------------------
// Projection: Vt[b,h,f,n] = (inp[b] @ Wk[h])[n,f] + bk[h,f]   (f16 out)
// One wave per 16x16 output tile; K-loop of 8 WMMAs over NI=256.
__global__ void proj_kernel(const _Float16* __restrict__ Xh,
                            const _Float16* __restrict__ Wt,
                            const float*    __restrict__ bk,
                            _Float16*       __restrict__ Vt) {
    int wave = (blockIdx.x * blockDim.x + threadIdx.x) >> 5;
    int lane = threadIdx.x & 31;
    int ft = wave & (NF / 16 - 1);          // 16 f-tiles
    int t  = wave >> 4;
    int nt = t & (N / 16 - 1);              // 64 n-tiles
    int bh = t >> 6;                        // 0..B*H-1
    int b = bh >> 2, h = bh & 3;

    int m    = lane & 15;                   // A row / B column within tile
    int half = lane >> 4;

    const _Float16* arow = Xh + (size_t)(b * N + nt * 16 + m) * NI;
    const _Float16* brow = Wt + (size_t)(h * NF + ft * 16 + m) * NI;

    v8f c = {};
    for (int kk = 0; kk < NI; kk += 32) {
        AF a, bfr;
        // A fragment (16x32 f16): lanes 0-15 -> K 0-7 & 16-23; lanes 16-31 -> K 8-15 & 24-31
        int oa = kk + half * 8;
        a.p.lo = *(const v8h*)(arow + oa);
        a.p.hi = *(const v8h*)(arow + oa + 16);
        // B fragment (32x16 f16): lanes 0-15 -> K 0-7 & 16-23; lanes 16-31 -> K 8-15 & 24-31
        int ob = kk + half * 8;
        bfr.p.lo = *(const v8h*)(brow + ob);
        bfr.p.hi = *(const v8h*)(brow + ob + 16);
        c = __builtin_amdgcn_wmma_f32_16x16x32_f16(false, a.v, false, bfr.v,
                                                   (short)0, c, false, false);
    }
    // Epilogue: lane holds column f, rows n = nt*16 + half*8 + r (contiguous) -> b128 store
    int f = ft * 16 + m;
    float bias = bk[h * NF + f];
    HF8 o;
#pragma unroll
    for (int r = 0; r < 8; ++r) o.h[r] = (_Float16)(c[r] + bias);
    _Float16* dst = Vt + ((size_t)(b * H + h) * NF + f) * N + nt * 16 + half * 8;
    *(v8h*)dst = o.v;
}

// ---------------------------------------------------------------------------
// T[b,h,n] = sum_f Vt[b,h,f,n] * a_dst[h,f]
__global__ void tproj_kernel(const _Float16* __restrict__ Vt,
                             const float*    __restrict__ a_dst,
                             float*          __restrict__ T) {
    int idx = blockIdx.x * 256 + threadIdx.x;   // over B*H*N
    int n  = idx & (N - 1);
    int bh = idx >> 10;
    int h  = bh & 3;
    const _Float16* col = Vt + (size_t)bh * NF * N + n;
    const float* ad = a_dst + h * NF;
    float t = 0.f;
    for (int f = 0; f < NF; ++f) t = fmaf((float)col[(size_t)f * N], ad[f], t);
    T[idx] = t;
}

// ---------------------------------------------------------------------------
// Fused masked softmax + aggregation.
// Block = 256 threads = 8 waves, handles one (b,h) x 16-row tile.
// Wave w owns output columns [w*32, w*32+32) -> two 16x16 f32 accumulators.
// P-tiles (attention probabilities in A-fragment layout) are computed ONCE per
// block (one chunk per wave) and shared through LDS, cutting exp/adj/T work 8x.
__global__ void attn_kernel(const int*      __restrict__ adj,
                            const float*    __restrict__ T,
                            const _Float16* __restrict__ Vt,
                            float*          __restrict__ out) {
    int it = blockIdx.x & (N / 16 - 1);
    int bh = blockIdx.x >> 6;
    int b = bh >> 2, h = bh & 3;
    int i0 = it * 16;
    const float* Trow = T + (size_t)bh * N;

    __shared__ float red[256];
    __shared__ float s_tmax[16];
    __shared__ float s_invd[16];
    // P staging: 8 chunks x 32 lanes x 16 halves (A-fragment element order) = 8 KB
    __shared__ __align__(32) _Float16 sP[8][32][16];

    int tid = threadIdx.x;
    int row = tid & 15, seg = tid >> 4;           // 16 threads per row
    const int* arow = adj + (size_t)(i0 + row) * N;

    // Pass 1: masked row max of T over neighbors
    float pm = -3.0e38f;
    for (int j = seg * 64; j < seg * 64 + 64; ++j)
        if (arow[j] > 0) pm = fmaxf(pm, Trow[j]);
    red[tid] = pm;
    __syncthreads();
    if (tid < 16) {
        float mx = -3.0e38f;
        for (int s = 0; s < 16; ++s) mx = fmaxf(mx, red[s * 16 + tid]);
        s_tmax[tid] = mx;
    }
    __syncthreads();
    // Pass 2: masked denom
    float tm = s_tmax[row];
    float ps = 0.f;
    for (int j = seg * 64; j < seg * 64 + 64; ++j)
        if (arow[j] > 0) ps += __expf(Trow[j] - tm);
    red[tid] = ps;
    __syncthreads();
    if (tid < 16) {
        float d = 0.f;
        for (int s = 0; s < 16; ++s) d += red[s * 16 + tid];
        s_invd[tid] = (d > 0.f) ? (1.0f / d) : 0.f;
    }
    __syncthreads();

    // Pass 3: WMMA aggregation over j, 8 chunks of 32 per LDS group
    int lane = tid & 31, wv = tid >> 5;
    int m = lane & 15, half = lane >> 4;          // A-fragment: lane -> row i
    const int* amrow = adj + (size_t)(i0 + m) * N;
    float tmax_m = s_tmax[m];
    float invd_m = s_invd[m];
    int fb = wv * 32;
    const _Float16* b0 = Vt + ((size_t)bh * NF + fb + m) * N;        // B col = f
    const _Float16* b1 = Vt + ((size_t)bh * NF + fb + 16 + m) * N;

    v8f c0 = {};
    v8f c1 = {};
    for (int jb = 0; jb < N; jb += 256) {
        // Producer: wave wv computes P for chunk jb + wv*32 (all 16 rows x 32 cols)
        {
            int j0 = jb + wv * 32;
#pragma unroll
            for (int v = 0; v < 8; ++v) {
                // A-fragment K index for VGPR v, this lane half
                int k0 = ((v & 3) << 1) + (half << 3) + ((v >> 2) << 4);
                int j = j0 + k0;
                float p0 = (amrow[j]     > 0) ? __expf(Trow[j]     - tmax_m) * invd_m : 0.f;
                float p1 = (amrow[j + 1] > 0) ? __expf(Trow[j + 1] - tmax_m) * invd_m : 0.f;
                sP[wv][lane][2 * v]     = (_Float16)p0;
                sP[wv][lane][2 * v + 1] = (_Float16)p1;
            }
        }
        __syncthreads();
        // Consumer: every wave runs all 8 chunks through its two accumulators
#pragma unroll 2
        for (int cck = 0; cck < 8; ++cck) {
            AF pa;
            pa.p.lo = *(const v8h*)&sP[cck][lane][0];
            pa.p.hi = *(const v8h*)&sP[cck][lane][8];
            int ob = jb + cck * 32 + half * 8;     // B-fragment: K 8h..8h+7 & 16+8h..16+8h+7 per lane
            AF bf0, bf1;
            bf0.p.lo = *(const v8h*)(b0 + ob);
            bf0.p.hi = *(const v8h*)(b0 + ob + 16);
            bf1.p.lo = *(const v8h*)(b1 + ob);
            bf1.p.hi = *(const v8h*)(b1 + ob + 16);
            c0 = __builtin_amdgcn_wmma_f32_16x16x32_f16(false, pa.v, false, bf0.v,
                                                        (short)0, c0, false, false);
            c1 = __builtin_amdgcn_wmma_f32_16x16x32_f16(false, pa.v, false, bf1.v,
                                                        (short)0, c1, false, false);
        }
        __syncthreads();
    }

    // Epilogue: out[b, n, h*NF + f]  (head concat)
#pragma unroll
    for (int r = 0; r < 8; ++r) {
        int n = i0 + half * 8 + r;
        size_t o = ((size_t)(b * N + n)) * (H * NF) + h * NF;
        out[o + fb + m]      = c0[r];
        out[o + fb + 16 + m] = c1[r];
    }
}

// ---------------------------------------------------------------------------
extern "C" void kernel_launch(void* const* d_in, const int* in_sizes, int n_in,
                              void* d_out, int out_size, void* d_ws, size_t ws_size,
                              hipStream_t stream) {
    (void)in_sizes; (void)n_in; (void)out_size; (void)ws_size;
    const float* inp   = (const float*)d_in[0];
    const int*   adj   = (const int*)d_in[1];
    const float* Wk    = (const float*)d_in[2];
    const float* bk    = (const float*)d_in[3];
    // d_in[4] = a_src, d_in[6] = a_b: cancel inside the row softmax, unused.
    const float* a_dst = (const float*)d_in[5];
    float* out = (float*)d_out;

    // Workspace carve-up (all sizes multiples of 256B; hipMalloc base is 256B aligned)
    char* ws = (char*)d_ws;
    const size_t XH_BYTES = (size_t)B * N * NI * 2;          // 16 MB
    const size_t WT_BYTES = (size_t)H * NF * NI * 2;         // 512 KB
    const size_t VT_BYTES = (size_t)B * H * NF * N * 2;      // 64 MB
    _Float16* Xh = (_Float16*)ws;
    _Float16* Wt = (_Float16*)(ws + XH_BYTES);
    _Float16* Vt = (_Float16*)(ws + XH_BYTES + WT_BYTES);
    float*    T  = (float*)(ws + XH_BYTES + WT_BYTES + VT_BYTES);

    cvt_x_kernel<<<(B * N * NI) / 256, 256, 0, stream>>>(inp, Xh);
    cvt_w_kernel<<<(H * NF * NI) / 256, 256, 0, stream>>>(Wk, Wt);

    // B*H * (N/16) * (NF/16) waves, 8 waves per block
    int proj_waves = B * H * (N / 16) * (NF / 16);
    proj_kernel<<<proj_waves / 8, 256, 0, stream>>>(Xh, Wt, bk, Vt);

    tproj_kernel<<<(B * H * N) / 256, 256, 0, stream>>>(Vt, a_dst, T);

    attn_kernel<<<B * H * (N / 16), 256, 0, stream>>>(adj, T, Vt, out);
}
